// MultiScaleAttention_76227079569740
// MI455X (gfx1250) — matrix operand order repaired and checked by reading, not compiled, not run on hardware
//
#include <hip/hip_runtime.h>
#include <stdint.h>

// ---- problem constants (from reference setup_inputs) ----
#define NH   8
#define PNT  8
#define CD   256
#define HH   64
#define WW   64
#define HWSZ 4096
#define BB   4
#define HF   128
#define WF   128

// ---- vector types (plain ext vectors: POD, union-safe) ----
typedef __bf16        v16bf __attribute__((ext_vector_type(16)));
typedef float         v8f   __attribute__((ext_vector_type(8)));
typedef unsigned int  u32x4 __attribute__((ext_vector_type(4)));
typedef int           i32x4 __attribute__((ext_vector_type(4)));
typedef int           i32x8 __attribute__((ext_vector_type(8)));
typedef float         f32x4 __attribute__((ext_vector_type(4)));

union FragU { u32x4 q[2]; v16bf v; };

__device__ __forceinline__ float bf2f(unsigned short h) {
  union { unsigned int u; float f; } c; c.u = ((unsigned int)h) << 16; return c.f;
}
__device__ __forceinline__ unsigned short f2bf(float f) {
  union { float f; unsigned int u; } c; c.f = f;
  unsigned int u = c.u;
  u += 0x7FFFu + ((u >> 16) & 1u);            // round-to-nearest-even
  return (unsigned short)(u >> 16);
}
__device__ __forceinline__ unsigned int pk2(float lo, float hi) {
  return (unsigned int)f2bf(lo) | ((unsigned int)f2bf(hi) << 16);
}

// =====================================================================
// Kernel 1: convert q_w / k_w / v_w / proj_w (all [256 out][256 in] f32)
// to bf16 in workspace. [out][in] row-major == WMMA B-fragment friendly.
// =====================================================================
__global__ void k_cvt_weights(const float* __restrict__ qw,
                              const float* __restrict__ kw,
                              const float* __restrict__ vw,
                              const float* __restrict__ pw,
                              unsigned short* __restrict__ dst) {
  int i = blockIdx.x * 256 + threadIdx.x;     // 0 .. 262143
  const float* src;
  int m = i >> 16, r = i & 65535;
  src = (m == 0) ? qw : (m == 1) ? kw : (m == 2) ? vw : pw;
  dst[i] = f2bf(src[r]);
}

// =====================================================================
// Kernel 2: features NCHW f32 -> NHWC bf16 (contiguous channels so each
// bilinear tap is a 512B contiguous read instead of 256 cachelines).
// One 32-lane group per pixel; lane handles 8 channels -> 16B store.
// =====================================================================
__global__ void k_feat_nhwc(const float* __restrict__ feats,
                            unsigned short* __restrict__ featsT) {
  int t = threadIdx.x;
  int pix  = blockIdx.x * 8 + (t >> 5);       // 0 .. 65535  (b*16384 + y*128 + x)
  int lane = t & 31;
  int b = pix >> 14, yx = pix & 16383;
  const float* src = feats + (size_t)b * CD * HF * WF + yx;   // stride HF*WF per ch
  u32x4 pk;
  #pragma unroll
  for (int j = 0; j < 4; ++j) {
    float lo = src[(size_t)(lane * 8 + 2 * j)     * (HF * WF)];
    float hi = src[(size_t)(lane * 8 + 2 * j + 1) * (HF * WF)];
    pk[j] = pk2(lo, hi);
  }
  *(u32x4*)(featsT + (size_t)pix * CD + lane * 8) = pk;
}

// =====================================================================
// Kernel 3: 3x3 offset conv + sampling coords + bilinear gather.
// One WG per 8x8 spatial tile (256 WGs). Query tile + weights staged in
// LDS as bf16 (4 channel phases of 64 ch -> < 64KB static LDS).
// Then one wave per (loc, p) sample: 4 taps x 512B contiguous reads,
// blended in f32, written as bf16 to points[b][loc][p][c].
// =====================================================================
__global__ void __launch_bounds__(256)
k_offset_sample(const float* __restrict__ query,
                const float* __restrict__ offw,
                const float* __restrict__ offb,
                const unsigned short* __restrict__ featsT,
                unsigned short* __restrict__ points) {
  __shared__ unsigned short sq[64 * 100];      // 64 ch x 10x10 tile (bf16)
  __shared__ unsigned short sw[16 * 64 * 9];   // 16 oc x 64 ch x 9 taps (bf16)
  __shared__ float part[4][64][16];            // partial conv sums
  __shared__ float convout[64 * 16];
  __shared__ float coords[64 * 8 * 2];

  int t   = threadIdx.x;
  int blk = blockIdx.x;                        // 256 = B * 8 * 8 tiles
  int b   = blk >> 6;
  int ty  = (blk >> 3) & 7, tx = blk & 7;
  int y0  = ty * 8, x0 = tx * 8;

  int loc = t & 63, csub = t >> 6;
  int ly  = loc >> 3, lx = loc & 7;

  float acc[16];
  #pragma unroll
  for (int i = 0; i < 16; ++i) acc[i] = 0.f;

  for (int ph = 0; ph < 4; ++ph) {
    // stage 64-channel slab of the 10x10 query tile (zero padded)
    for (int e = t; e < 6400; e += 256) {
      int c = e / 100, r = e % 100, yy = r / 10, xx = r % 10;
      int gy = y0 + yy - 1, gx = x0 + xx - 1;
      float v = 0.f;
      if (gy >= 0 && gy < HH && gx >= 0 && gx < WW)
        v = query[(((size_t)b * CD + ph * 64 + c) * HH + gy) * WW + gx];
      sq[e] = f2bf(v);
    }
    // stage matching weight slab
    for (int e = t; e < 9216; e += 256) {
      int oc = e / 576, r = e % 576, c = r / 9, tap = r % 9;
      sw[e] = f2bf(offw[((size_t)oc * CD + ph * 64 + c) * 9 + tap]);
    }
    __syncthreads();
    for (int cl = 0; cl < 16; ++cl) {
      int c = csub * 16 + cl;
      #pragma unroll
      for (int ky = 0; ky < 3; ++ky)
        #pragma unroll
        for (int kx = 0; kx < 3; ++kx) {
          float qv = bf2f(sq[c * 100 + (ly + ky) * 10 + (lx + kx)]);
          int tap = ky * 3 + kx;
          #pragma unroll
          for (int oc = 0; oc < 16; ++oc)
            acc[oc] += qv * bf2f(sw[(oc * 64 + c) * 9 + tap]);
        }
    }
    __syncthreads();
  }
  #pragma unroll
  for (int oc = 0; oc < 16; ++oc) part[csub][loc][oc] = acc[oc];
  __syncthreads();

  // reduce channel subsets + bias
  for (int i = t; i < 1024; i += 256) {
    int l = i >> 4, oc = i & 15;
    convout[l * 16 + oc] = part[0][l][oc] + part[1][l][oc] +
                           part[2][l][oc] + part[3][l][oc] + offb[oc];
  }
  __syncthreads();

  // sampling coordinates; reference stacks ref as (y,x) but grid_sample
  // reads grid[...,0] as x -> channel 2p (the /H one) becomes grid-x.
  for (int s = t; s < 512; s += 256) {
    int l = s >> 3, p = s & 7;
    float offy = convout[l * 16 + 2 * p];
    float offx = convout[l * 16 + 2 * p + 1];
    int gy = y0 + (l >> 3), gx = x0 + (l & 7);
    float refy = -1.f + 2.f * (float)gy / (float)(HH - 1);
    float refx = -1.f + 2.f * (float)gx / (float)(WW - 1);
    coords[s * 2 + 0] = offy / (float)HH + refy;   // grid x
    coords[s * 2 + 1] = offx / (float)WW + refx;   // grid y
  }
  __syncthreads();

  // one wave per sample: lane owns 8 channels
  int wid = t >> 5, lane = t & 31;
  for (int s = wid; s < 512; s += 8) {
    int l = s >> 3, p = s & 7;
    float xg = coords[s * 2 + 0], yg = coords[s * 2 + 1];
    float x = (xg + 1.f) * (WF * 0.5f) - 0.5f;
    float y = (yg + 1.f) * (HF * 0.5f) - 0.5f;
    float x0f = floorf(x), y0f = floorf(y);
    float txf = x - x0f, tyf = y - y0f;
    int xi = (int)x0f, yi = (int)y0f;
    float fa[8];
    #pragma unroll
    for (int j = 0; j < 8; ++j) fa[j] = 0.f;
    #pragma unroll
    for (int tap = 0; tap < 4; ++tap) {
      int xs = xi + (tap & 1), ys = yi + (tap >> 1);
      float w = ((tap & 1) ? txf : 1.f - txf) * ((tap >> 1) ? tyf : 1.f - tyf);
      bool valid = (xs >= 0) && (xs < WF) && (ys >= 0) && (ys < HF);
      w = valid ? w : 0.f;
      if (w != 0.f) {                         // wave-uniform branch
        int xc = min(max(xs, 0), WF - 1), yc = min(max(ys, 0), HF - 1);
        u32x4 v = *(const u32x4*)(featsT +
                    (((size_t)b * HF + yc) * WF + xc) * CD + lane * 8);
        #pragma unroll
        for (int j = 0; j < 4; ++j) {
          fa[2 * j]     += w * bf2f((unsigned short)(v[j] & 0xffffu));
          fa[2 * j + 1] += w * bf2f((unsigned short)(v[j] >> 16));
        }
      }
    }
    u32x4 pk;
    #pragma unroll
    for (int j = 0; j < 4; ++j) pk[j] = pk2(fa[2 * j], fa[2 * j + 1]);
    int gl = (y0 + (l >> 3)) * WW + (x0 + (l & 7));
    *(u32x4*)(points + (((size_t)b * HWSZ + gl) * PNT + p) * CD + lane * 8) = pk;
  }
}

// =====================================================================
// WMMA fragment loaders (bf16 16x16x32; A and B share one contiguous k order):
//   A: lane -> M row = lane%16; 16 contiguous K at (lane/16)*16.
//   B: lane -> N col = lane%16; 16 contiguous K at (lane/16)*16.
//   D: lane l, vgpr r -> row r + 8*(l/16), col l%16.
// =====================================================================
#define ASTRIDE 264   // bf16 row stride for A tiles in LDS (16B aligned, bank-spread)
#define KVSTRIDE 132  // f32 m-stride for k/v result in LDS

__device__ __forceinline__ v16bf load_afrag(const unsigned short* base,
                                            int lane, int mtile, int k0) {
  int row = mtile * 16 + (lane & 15);
  int kb  = k0 * 32 + (lane >> 4) * 16;
  FragU f;
  f.q[0] = *(const u32x4*)(base + row * ASTRIDE + kb);
  f.q[1] = *(const u32x4*)(base + row * ASTRIDE + kb + 8);
  return f.v;
}
__device__ __forceinline__ v16bf load_bfrag(const unsigned short* __restrict__ w,
                                            int lane, int ntile, int k0) {
  int n  = ntile * 16 + (lane & 15);
  int kb = k0 * 32 + (lane >> 4) * 16;
  const u32x4* p = (const u32x4*)(w + n * 256 + kb);
  FragU f;
  f.q[0] = p[0];
  f.q[1] = p[1];
  return f.v;
}
#define WMMA_BF16(a, b, c) \
  __builtin_amdgcn_wmma_f32_16x16x32_bf16(false, (a), false, (b), (short)0, (c), false, false)

// dynamic LDS layout for k_attn
#define OFF_PTS 0                          // [128][264] bf16  =  67584
#define OFF_KV  67584                      // [256][132] f32   = 135168
#define OFF_Q   202752                     // [16][264]  f32   =  16896
#define OFF_AT  219648                     // [16][64]   f32   =   4096
#define OFF_OP  223744                     // [16][264]  bf16  =   8448
#define OFF_AQ  232192                     // [16][264]  bf16  =   8448
#define SMEM_ATTN 240640                   // 235 KB of the 320 KB WGP LDS

// =====================================================================
// Kernel 4: fused q/k/v projections + softmax attention + output proj.
// One WG (8 waves) per 16 consecutive locations; all GEMMs via
// v_wmma_f32_16x16x32_bf16 with f32 accumulation. Points tile is staged
// through the Tensor Data Mover: one 2-D descriptor (256x128 elems,
// data_size=2B) with HW LDS padding (pad_interval=128 dwords -> row of
// 512B, pad_amount=4 dwords -> 16B) reproducing the 528B bank-padded
// row stride with zero VALU/DS work.
// =====================================================================
__global__ void __launch_bounds__(256)
k_attn(const float* __restrict__ query,
       const unsigned short* __restrict__ pts_g,
       const unsigned short* __restrict__ wq,
       const unsigned short* __restrict__ wk,
       const unsigned short* __restrict__ wvm,
       const unsigned short* __restrict__ wp,
       const float* __restrict__ qb, const float* __restrict__ kb,
       const float* __restrict__ vb, const float* __restrict__ pb,
       float* __restrict__ out) {
  extern __shared__ char smem[];
  unsigned short* sp  = (unsigned short*)(smem + OFF_PTS); // points tile  [128][264]
  float*          skv = (float*)(smem + OFF_KV);           // k (then v)   [n=256][m=132]
  float*          sq  = (float*)(smem + OFF_Q);            // q result     [16][264]
  float*          sat = (float*)(smem + OFF_AT);           // logits/attn  [16][8][8]
  unsigned short* sop = (unsigned short*)(smem + OFF_OP);  // pre-proj out [16][264]
  unsigned short* saq = (unsigned short*)(smem + OFF_AQ);  // query tile A [16][264]

  int t = threadIdx.x, lane = t & 31, wid = t >> 5;
  int g0   = blockIdx.x * 16;                  // 1024 blocks, tiles never straddle b
  int b    = g0 >> 12;
  int loc0 = g0 & 4095;

  // pull upcoming weight streams toward L2/L0 (global_prefetch_b8)
  __builtin_prefetch(wq + (size_t)t * 256, 0, 1);
  __builtin_prefetch(wk + (size_t)t * 256, 0, 1);

  // ---- stage points tile via TDM: async DMA with hardware LDS padding ----
  if (wid == 0) {
    unsigned long long gaddr =
        (unsigned long long)(pts_g + (size_t)g0 * PNT * CD);
    u32x4 g0d;
    g0d[0] = 1u;                                   // count=1, user-mode D#
    g0d[1] = (unsigned int)OFF_PTS;                // lds_addr (bytes)
    g0d[2] = (unsigned int)(gaddr & 0xffffffffu);  // global_addr[31:0]
    g0d[3] = (unsigned int)((gaddr >> 32) & 0x01ffffffu) | (2u << 30); // type=2
    i32x8 g1d;
    g1d[0] = (int)((1u << 16)      // data_size = 2 bytes
                 | (1u << 20)      // pad_enable
                 | (6u << 22)      // pad_interval: 2^(6+1)=128 dwords (512B row)
                 | (3u << 25));    // pad_amount: 3+1 = 4 dwords (16B)
    g1d[1] = (int)(256u << 16);    // tensor_dim0 = 256 elems (bits 79:48)
    g1d[2] = (int)(128u << 16);    // tensor_dim1 = 128 rows  (bits 111:80)
    g1d[3] = (int)(256u << 16);    // tile_dim0   = 256 elems (bits 127:112)
    g1d[4] = 128;                  // tile_dim1   = 128 rows  (bits 143:128)
    g1d[5] = 256;                  // tensor_dim0_stride = 256 elems (bits 207:160)
    g1d[6] = 0;
    g1d[7] = 0;
    i32x4 z4 = {0, 0, 0, 0};
#if defined(__clang_major__) && (__clang_major__ >= 23)
    i32x8 z8 = {0, 0, 0, 0, 0, 0, 0, 0};
    __builtin_amdgcn_tensor_load_to_lds(g0d, g1d, z4, z4, z8, 0);
#else
    __builtin_amdgcn_tensor_load_to_lds(g0d, g1d, z4, z4, 0);
#endif
    __builtin_amdgcn_s_wait_tensorcnt(0);
  }

  // ---- stage query tile as bf16 A (thread t = channel t, 16 locs) ----
  {
    const f32x4* q4 = (const f32x4*)(query + ((size_t)b * CD + t) * HWSZ + loc0);
    #pragma unroll
    for (int mm = 0; mm < 4; ++mm) {
      f32x4 v = q4[mm];
      #pragma unroll
      for (int j = 0; j < 4; ++j) saq[(mm * 4 + j) * ASTRIDE + t] = f2bf(v[j]);
    }
  }
  __syncthreads();

  // ---- q GEMM: [16,256] = saq x wq^T ; 16 N-tiles over 8 waves ----
  {
    v8f a0 = {}, a1 = {};
    int nt0 = wid * 2, nt1 = wid * 2 + 1;
    for (int k0 = 0; k0 < 8; ++k0) {
      v16bf a  = load_afrag(saq, lane, 0, k0);
      v16bf b0 = load_bfrag(wq, lane, nt0, k0);
      v16bf b1 = load_bfrag(wq, lane, nt1, k0);
      a0 = WMMA_BF16(a, b0, a0);
      a1 = WMMA_BF16(a, b1, a1);
    }
    #pragma unroll
    for (int j = 0; j < 2; ++j) {
      v8f acc = j ? a1 : a0;
      int n = (wid * 2 + j) * 16 + (lane & 15);
      float bias = qb[n];
      #pragma unroll
      for (int r = 0; r < 8; ++r)
        sq[(r + 8 * (lane >> 4)) * ASTRIDE + n] = acc[r] + bias;
    }
  }
  __syncthreads();

  // ---- k GEMM: [128,256] = pts x wk^T ; wave owns one M-tile ----
  {
    v8f acc[16] = {};
    for (int k0 = 0; k0 < 8; ++k0) {
      v16bf a = load_afrag(sp, lane, wid, k0);
      #pragma unroll
      for (int nt = 0; nt < 16; ++nt) {
        v16bf bm = load_bfrag(wk, lane, nt, k0);
        acc[nt] = WMMA_BF16(a, bm, acc[nt]);
      }
    }
    #pragma unroll
    for (int nt = 0; nt < 16; ++nt) {
      int n = nt * 16 + (lane & 15);
      float bias = kb[n];
      int m0 = wid * 16 + 8 * (lane >> 4);
      f32x4 lo = {acc[nt][0] + bias, acc[nt][1] + bias, acc[nt][2] + bias, acc[nt][3] + bias};
      f32x4 hi = {acc[nt][4] + bias, acc[nt][5] + bias, acc[nt][6] + bias, acc[nt][7] + bias};
      *(f32x4*)(skv + n * KVSTRIDE + m0)     = lo;
      *(f32x4*)(skv + n * KVSTRIDE + m0 + 4) = hi;
    }
  }
  __syncthreads();

  // ---- logits: q . k per (loc, head, point), then softmax over P ----
  const float scale = 0.17677669529663687f;   // 32^-0.5
  #pragma unroll
  for (int i = 0; i < 4; ++i) {
    int idx = t + 256 * i;                    // 16*8*8 = 1024 combos
    int l = idx >> 6, h = (idx >> 3) & 7, p = idx & 7;
    const float* qrow = sq  + l * ASTRIDE + h * 32;
    const float* krow = skv + (h * 32) * KVSTRIDE + (l * 8 + p);
    float s = 0.f;
    #pragma unroll
    for (int d = 0; d < 32; ++d) s += qrow[d] * krow[d * KVSTRIDE];
    sat[l * 64 + h * 8 + p] = s * scale;
  }
  __syncthreads();
  if (t < 128) {
    float* a = sat + (t >> 3) * 64 + (t & 7) * 8;
    float mx = a[0];
    #pragma unroll
    for (int p = 1; p < 8; ++p) mx = fmaxf(mx, a[p]);
    float e[8], sum = 0.f;
    #pragma unroll
    for (int p = 0; p < 8; ++p) { e[p] = __expf(a[p] - mx); sum += e[p]; }
    float inv = 1.f / sum;
    #pragma unroll
    for (int p = 0; p < 8; ++p) a[p] = e[p] * inv;
  }
  __syncthreads();

  // ---- v GEMM: same shape as k, overwrites skv ----
  {
    v8f acc[16] = {};
    for (int k0 = 0; k0 < 8; ++k0) {
      v16bf a = load_afrag(sp, lane, wid, k0);
      #pragma unroll
      for (int nt = 0; nt < 16; ++nt) {
        v16bf bm = load_bfrag(wvm, lane, nt, k0);
        acc[nt] = WMMA_BF16(a, bm, acc[nt]);
      }
    }
    #pragma unroll
    for (int nt = 0; nt < 16; ++nt) {
      int n = nt * 16 + (lane & 15);
      float bias = vb[n];
      int m0 = wid * 16 + 8 * (lane >> 4);
      f32x4 lo = {acc[nt][0] + bias, acc[nt][1] + bias, acc[nt][2] + bias, acc[nt][3] + bias};
      f32x4 hi = {acc[nt][4] + bias, acc[nt][5] + bias, acc[nt][6] + bias, acc[nt][7] + bias};
      *(f32x4*)(skv + n * KVSTRIDE + m0)     = lo;
      *(f32x4*)(skv + n * KVSTRIDE + m0 + 4) = hi;
    }
  }
  __syncthreads();

  // ---- attn-weighted sum over P -> pre-proj tile (bf16 A for proj GEMM) ----
  #pragma unroll
  for (int i = 0; i < 16; ++i) {
    int idx = t + 256 * i;                    // 16*256 outputs
    int l = idx >> 8, c = idx & 255;
    const float* a  = sat + l * 64 + (c >> 5) * 8;
    const float* vv = skv + c * KVSTRIDE + l * 8;
    float s = 0.f;
    #pragma unroll
    for (int p = 0; p < 8; ++p) s += a[p] * vv[p];
    sop[l * ASTRIDE + c] = f2bf(s);
  }
  __syncthreads();

  // ---- proj GEMM + bias, store NCHW (coalesced float4 per lane) ----
  {
    v8f a0 = {}, a1 = {};
    for (int k0 = 0; k0 < 8; ++k0) {
      v16bf a  = load_afrag(sop, lane, 0, k0);
      v16bf b0 = load_bfrag(wp, lane, wid * 2, k0);
      v16bf b1 = load_bfrag(wp, lane, wid * 2 + 1, k0);
      a0 = WMMA_BF16(a, b0, a0);
      a1 = WMMA_BF16(a, b1, a1);
    }
    #pragma unroll
    for (int j = 0; j < 2; ++j) {
      v8f acc = j ? a1 : a0;
      int oc = (wid * 2 + j) * 16 + (lane & 15);
      float bias = pb[oc];
      float* dst = out + ((size_t)b * CD + oc) * HWSZ + loc0 + 8 * (lane >> 4);
      f32x4 lo = {acc[0] + bias, acc[1] + bias, acc[2] + bias, acc[3] + bias};
      f32x4 hi = {acc[4] + bias, acc[5] + bias, acc[6] + bias, acc[7] + bias};
      *(f32x4*)dst       = lo;
      *((f32x4*)dst + 1) = hi;
    }
  }
}

// =====================================================================
// launch
// workspace layout (bytes):
//   [0, 33554432)            featsT  NHWC bf16
//   [33554432, 100663296)    points  [B][HW][P][C] bf16
//   [100663296, 101187584)   bf16 weights: q_w, k_w, v_w, proj_w
// =====================================================================
extern "C" void kernel_launch(void* const* d_in, const int* in_sizes, int n_in,
                              void* d_out, int out_size, void* d_ws, size_t ws_size,
                              hipStream_t stream) {
  (void)in_sizes; (void)n_in; (void)out_size; (void)ws_size;
  const float* query = (const float*)d_in[0];
  const float* feats = (const float*)d_in[1];
  const float* offw  = (const float*)d_in[2];
  const float* offb  = (const float*)d_in[3];
  const float* qw    = (const float*)d_in[4];
  const float* qb    = (const float*)d_in[5];
  const float* kw    = (const float*)d_in[6];
  const float* kb    = (const float*)d_in[7];
  const float* vw    = (const float*)d_in[8];
  const float* vb    = (const float*)d_in[9];
  const float* pw    = (const float*)d_in[10];
  const float* pb    = (const float*)d_in[11];
  float* out = (float*)d_out;

  char* ws = (char*)d_ws;
  unsigned short* featsT = (unsigned short*)ws;
  unsigned short* points = (unsigned short*)(ws + 33554432);
  unsigned short* wbf    = (unsigned short*)(ws + 100663296);
  unsigned short* wqb = wbf;
  unsigned short* wkb = wbf + 65536;
  unsigned short* wvb = wbf + 131072;
  unsigned short* wpb = wbf + 196608;

  k_cvt_weights<<<1024, 256, 0, stream>>>(qw, kw, vw, pw, wbf);
  k_feat_nhwc<<<8192, 256, 0, stream>>>(feats, featsT);
  k_offset_sample<<<256, 256, 0, stream>>>(query, offw, offb, featsT, points);
  k_attn<<<1024, 256, SMEM_ATTN, stream>>>(query, points, wqb, wkb, wvb, wpb,
                                           qb, kb, vb, pb, out);
}
